// NeuromorphicBrain_15195594293715
// MI455X (gfx1250) — matrix operand order repaired and checked by reading, not compiled, not run on hardware
//
#include <hip/hip_runtime.h>
#include <math.h>

// ---------------------------------------------------------------------------
// NeuromorphicBrain forward pass for gfx1250 (MI455X), wave32 + WMMA bf16.
//
//  - repack_kernel: f32 weights -> bf16 fragments pre-swizzled into the
//    V_WMMA_F32_16X16X32_BF16 B-matrix lane layout (zero padded to K%32,
//    N%16). Self-recurrent cells fold Wi[h-rows]+Wh into one block;
//    exp(log_step)*dt precomputed.
//  - brain_scan_kernel: 16 WGs (one per 16-row batch tile) x 8 waves.
//    T=512 recurrence in-kernel; all matmuls via WMMA bf16/f32-acc;
//    activations in LDS bf16 (padded cols zeroed); f32 LDS staging for
//    cell update + layernorm; wave-uniform (readfirstlane) job dispatch so
//    WMMA sits under scalar branches with EXEC all-ones; parallel shuffle
//    reductions for layernorm stats; prefetch of next x tile.
//  - head kernels: predictions+surprise, anomaly, properties.
// ---------------------------------------------------------------------------

typedef float  v8f   __attribute__((ext_vector_type(8)));
typedef __bf16 v16bf __attribute__((ext_vector_type(16)));
typedef __bf16 v8bf  __attribute__((ext_vector_type(8)));

constexpr int TSTEPS = 512, DIN = 32, DH = 256;
constexpr int NTHR = 256;  // 8 waves

// ---- output layout (flat f32, reference return order) ----
constexpr long long PRED_OFF = 0;
constexpr long long HID_OFF  = 256LL * 511 * 32;
constexpr long long ANOM_OFF = HID_OFF + 256LL * 512 * 256;
constexpr long long SURP_OFF = ANOM_OFF + 256LL * 512;
constexpr long long PROP_OFF = SURP_OFF + 256LL * 511;

// ---- packed weight-block offsets (in __bf16 units, inside d_ws) ----
constexpr int OFF_IP     = 0;
constexpr int OFF_H2S_G  = OFF_IP     + 32 * 64;
constexpr int OFF_H2S_P  = OFF_H2S_G  + 64 * 64;
constexpr int OFF_SEN_I  = OFF_H2S_P  + 64 * 64;
constexpr int OFF_SEN_H  = OFF_SEN_I  + 64 * 64;
constexpr int OFF_SEN_GI = OFF_SEN_H  + 64 * 64;
constexpr int OFF_SEN_GH = OFF_SEN_GI + 64 * 64;
constexpr int OFF_HIP_CI = OFF_SEN_GH + 64 * 64;
constexpr int OFF_HIP_CH = OFF_HIP_CI + 64 * 64;  // Wi[64:115] + Wh folded
constexpr int OFF_HIP_GI = OFF_HIP_CH + 64 * 64;
constexpr int OFF_HIP_GH = OFF_HIP_GI + 64 * 64;
constexpr int OFF_CER_CI = OFF_HIP_GH + 64 * 64;
constexpr int OFF_CER_CP = OFF_CER_CI + 64 * 64;  // pf part, K 90->96
constexpr int OFF_CER_CH = OFF_CER_CP + 96 * 64;
constexpr int OFF_CER_GI = OFF_CER_CH + 64 * 64;
constexpr int OFF_CER_GP = OFF_CER_GI + 64 * 64;
constexpr int OFF_CER_GH = OFF_CER_GP + 96 * 64;
constexpr int OFF_C2P_G  = OFF_CER_GH + 64 * 64;
constexpr int OFF_C2P_P  = OFF_C2P_G  + 64 * 96;
constexpr int OFF_PRE_CS = OFF_C2P_P  + 64 * 96;
constexpr int OFF_PRE_CH = OFF_PRE_CS + 64 * 96;
constexpr int OFF_PRE_CB = OFF_PRE_CH + 64 * 96;
constexpr int OFF_PRE_CF = OFF_PRE_CB + 64 * 96;  // Wi[166:256] + Wh folded
constexpr int OFF_PRE_GS = OFF_PRE_CF + 96 * 96;
constexpr int OFF_PRE_GH = OFF_PRE_GS + 64 * 96;
constexpr int OFF_PRE_GB = OFF_PRE_GH + 64 * 96;
constexpr int OFF_PRE_GF = OFF_PRE_GB + 64 * 96;
constexpr int OFF_THL_S  = OFF_PRE_GF + 96 * 96;
constexpr int OFF_THL_H  = OFF_THL_S  + 64 * 256;
constexpr int OFF_THL_P  = OFF_THL_H  + 64 * 256;
constexpr int OFF_THL_C  = OFF_THL_P  + 96 * 256;
constexpr int OFF_THP_S  = OFF_THL_C  + 64 * 256;
constexpr int OFF_THP_H  = OFF_THP_S  + 64 * 256;
constexpr int OFF_THP_P  = OFF_THP_H  + 64 * 256;
constexpr int OFF_THP_C  = OFF_THP_P  + 96 * 256;
constexpr int W_TOTAL    = OFF_THP_C  + 64 * 256;   // 286,720 halves
constexpr size_t VOFF_BYTES = (size_t)W_TOTAL * 2;

// ---- f32 vector region layout (indices into vbuf) ----
constexpr int V_B_IN   = 0;
constexpr int V_H2S_GB = 64,   V_H2S_PB = 128;
constexpr int V_SEN_B  = 192,  V_SEN_BG = 256,  V_SEN_ST = 320,  V_SEN_LG = 384,  V_SEN_LB = 448;
constexpr int V_HIP_B  = 512,  V_HIP_BG = 576,  V_HIP_ST = 640,  V_HIP_LG = 704,  V_HIP_LB = 768;
constexpr int V_CER_B  = 832,  V_CER_BG = 896,  V_CER_ST = 960,  V_CER_LG = 1024, V_CER_LB = 1088;
constexpr int V_C2P_GB = 1152, V_C2P_PB = 1248;
constexpr int V_PRE_B  = 1344, V_PRE_BG = 1440, V_PRE_ST = 1536, V_PRE_LG = 1632, V_PRE_LB = 1728;
constexpr int V_THL_B  = 1824, V_THP_B  = 2080, V_TH_LG  = 2336, V_TH_LB  = 2592;
constexpr int V_TOTAL  = 2848;

// ---- input pointer indices (setup_inputs dict order, depth-first) ----
struct P67 { const float* q[67]; };

// ===========================================================================
// repack kernel
// ===========================================================================
__device__ inline void packB(__bf16* wb, int off, int Kpad, int Npad,
                             const float* S1, int koff, const float* S2,
                             int Kr, int Nr) {
  const int NT = Npad >> 4;
  const int total = Kpad * Npad;
  for (int idx = threadIdx.x; idx < total; idx += blockDim.x) {
    int i    = idx & 15;
    int lane = (idx >> 4) & 31;
    int rest = idx >> 9;
    int nt = rest % NT, ks = rest / NT;
    int k = ks * 32 + ((lane & 16) ? 8 : 0) + (i < 8 ? i : i + 8);  // B frag: lane<16 -> K 0..7 & 16..23
    int n = nt * 16 + (lane & 15);
    float v = 0.f;
    if (k < Kr && n < Nr) {
      v = S1[(size_t)(k + koff) * Nr + n];
      if (S2) v += S2[(size_t)k * Nr + n];
    }
    wb[off + idx] = (__bf16)v;
  }
}

__global__ void __launch_bounds__(256)
repack_kernel(P67 P, __bf16* __restrict__ wb, float* __restrict__ vb) {
  const float* const* q = P.q;
  switch (blockIdx.x) {
    case 0:  packB(wb, OFF_IP,     32, 64, q[1],      0, nullptr,  32, 64);  break;
    case 1:  packB(wb, OFF_H2S_G,  64, 64, q[39 + 2], 0, nullptr,  51, 64);  break;
    case 2:  packB(wb, OFF_H2S_P,  64, 64, q[39 + 0], 0, nullptr,  51, 64);  break;
    case 3:  packB(wb, OFF_SEN_I,  64, 64, q[3 + 0],  0, nullptr,  64, 64);  break;
    case 4:  packB(wb, OFF_SEN_H,  64, 64, q[3 + 1],  0, nullptr,  64, 64);  break;
    case 5:  packB(wb, OFF_SEN_GI, 64, 64, q[3 + 3],  0, nullptr,  64, 64);  break;
    case 6:  packB(wb, OFF_SEN_GH, 64, 64, q[3 + 4],  0, nullptr,  64, 64);  break;
    case 7:  packB(wb, OFF_HIP_CI, 64, 64, q[12 + 0], 0,  nullptr,   64, 51); break;
    case 8:  packB(wb, OFF_HIP_CH, 64, 64, q[12 + 0], 64, q[12 + 1], 51, 51); break;
    case 9:  packB(wb, OFF_HIP_GI, 64, 64, q[12 + 3], 0,  nullptr,   64, 51); break;
    case 10: packB(wb, OFF_HIP_GH, 64, 64, q[12 + 3], 64, q[12 + 4], 51, 51); break;
    case 11: packB(wb, OFF_CER_CI, 64, 64, q[21 + 0], 0,  nullptr,  64, 51); break;
    case 12: packB(wb, OFF_CER_CP, 96, 64, q[21 + 0], 64, nullptr,  90, 51); break;
    case 13: packB(wb, OFF_CER_CH, 64, 64, q[21 + 1], 0,  nullptr,  51, 51); break;
    case 14: packB(wb, OFF_CER_GI, 64, 64, q[21 + 3], 0,  nullptr,  64, 51); break;
    case 15: packB(wb, OFF_CER_GP, 96, 64, q[21 + 3], 64, nullptr,  90, 51); break;
    case 16: packB(wb, OFF_CER_GH, 64, 64, q[21 + 4], 0,  nullptr,  51, 51); break;
    case 17: packB(wb, OFF_C2P_G,  64, 96, q[43 + 2], 0,  nullptr,  51, 90); break;
    case 18: packB(wb, OFF_C2P_P,  64, 96, q[43 + 0], 0,  nullptr,  51, 90); break;
    case 19: packB(wb, OFF_PRE_CS, 64, 96, q[30 + 0], 0,   nullptr,   64, 90); break;
    case 20: packB(wb, OFF_PRE_CH, 64, 96, q[30 + 0], 64,  nullptr,   51, 90); break;
    case 21: packB(wb, OFF_PRE_CB, 64, 96, q[30 + 0], 115, nullptr,   51, 90); break;
    case 22: packB(wb, OFF_PRE_CF, 96, 96, q[30 + 0], 166, q[30 + 1], 90, 90); break;
    case 23: packB(wb, OFF_PRE_GS, 64, 96, q[30 + 3], 0,   nullptr,   64, 90); break;
    case 24: packB(wb, OFF_PRE_GH, 64, 96, q[30 + 3], 64,  nullptr,   51, 90); break;
    case 25: packB(wb, OFF_PRE_GB, 64, 96, q[30 + 3], 115, nullptr,   51, 90); break;
    case 26: packB(wb, OFF_PRE_GF, 96, 96, q[30 + 3], 166, q[30 + 4], 90, 90); break;
    case 27: packB(wb, OFF_THL_S, 64, 256, q[47], 0,   nullptr, 64, 256); break;
    case 28: packB(wb, OFF_THL_H, 64, 256, q[47], 64,  nullptr, 51, 256); break;
    case 29: packB(wb, OFF_THL_P, 96, 256, q[47], 115, nullptr, 90, 256); break;
    case 30: packB(wb, OFF_THL_C, 64, 256, q[47], 205, nullptr, 51, 256); break;
    case 31: packB(wb, OFF_THP_S, 64, 256, q[51], 0,   nullptr, 64, 256); break;
    case 32: packB(wb, OFF_THP_H, 64, 256, q[51], 64,  nullptr, 51, 256); break;
    case 33: packB(wb, OFF_THP_P, 96, 256, q[51], 115, nullptr, 90, 256); break;
    case 34: packB(wb, OFF_THP_C, 64, 256, q[51], 205, nullptr, 51, 256); break;
    case 35: {
      int t = threadIdx.x;
      if (t < 64) {
        vb[V_B_IN + t]   = q[2][t];
        vb[V_H2S_GB + t] = q[39 + 3][t];
        vb[V_H2S_PB + t] = q[39 + 1][t];
        vb[V_SEN_B + t]  = q[3 + 2][t];
        vb[V_SEN_BG + t] = q[3 + 5][t];
        vb[V_SEN_ST + t] = __expf(q[3 + 6][t]) * 2.0f;
        vb[V_SEN_LG + t] = q[3 + 7][t];
        vb[V_SEN_LB + t] = q[3 + 8][t];
        bool r = t < 51;
        vb[V_HIP_B + t]  = r ? q[12 + 2][t] : 0.f;
        vb[V_HIP_BG + t] = r ? q[12 + 5][t] : 0.f;
        vb[V_HIP_ST + t] = r ? __expf(q[12 + 6][t]) * 0.8f : 0.f;
        vb[V_HIP_LG + t] = r ? q[12 + 7][t] : 0.f;
        vb[V_HIP_LB + t] = r ? q[12 + 8][t] : 0.f;
        vb[V_CER_B + t]  = r ? q[21 + 2][t] : 0.f;
        vb[V_CER_BG + t] = r ? q[21 + 5][t] : 0.f;
        vb[V_CER_ST + t] = r ? __expf(q[21 + 6][t]) * 1.8f : 0.f;
        vb[V_CER_LG + t] = r ? q[21 + 7][t] : 0.f;
        vb[V_CER_LB + t] = r ? q[21 + 8][t] : 0.f;
      }
      if (t < 96) {
        bool r = t < 90;
        vb[V_C2P_GB + t] = r ? q[43 + 3][t] : 0.f;
        vb[V_C2P_PB + t] = r ? q[43 + 1][t] : 0.f;
        vb[V_PRE_B + t]  = r ? q[30 + 2][t] : 0.f;
        vb[V_PRE_BG + t] = r ? q[30 + 5][t] : 0.f;
        vb[V_PRE_ST + t] = r ? __expf(q[30 + 6][t]) * 0.3f : 0.f;
        vb[V_PRE_LG + t] = r ? q[30 + 7][t] : 0.f;
        vb[V_PRE_LB + t] = r ? q[30 + 8][t] : 0.f;
      }
      if (t < 256) {
        vb[V_THL_B + t] = q[48][t];
        vb[V_THP_B + t] = q[52][t];
        vb[V_TH_LG + t] = q[49][t];
        vb[V_TH_LB + t] = q[50][t];
      }
      break;
    }
  }
}

// ===========================================================================
// WMMA fragment helpers (ISA 7.12.2 layouts)
// ===========================================================================
__device__ inline v16bf afrag(const __bf16* A, int lda, int ks, int lane) {
  // 16-bit A 16x32: lane<16 -> K 0..7 & 16..23 ; lane>=16 -> K 8..15 & 24..31
  int m = lane & 15;
  int koff = ks * 32 + ((lane & 16) ? 8 : 0);
  const __bf16* p = A + m * lda + koff;
  v8bf lo = *(const v8bf*)p;
  v8bf hi = *(const v8bf*)(p + 16);
  return __builtin_shufflevector(lo, hi, 0, 1, 2, 3, 4, 5, 6, 7,
                                 8, 9, 10, 11, 12, 13, 14, 15);
}

__device__ inline v8f gacc(v8f acc, const __bf16* A, int lda,
                           const __bf16* wb, int woff, int NTblk, int nt,
                           int ksteps, int lane) {
  for (int ks = 0; ks < ksteps; ++ks) {
    v16bf a = afrag(A, lda, ks, lane);
    v16bf b = *(const v16bf*)(wb + woff + (((ks * NTblk + nt) * 32 + lane) << 4));
    acc = __builtin_amdgcn_wmma_f32_16x16x32_bf16(false, a, false, b,
                                                  (short)0, acc, false, false);
  }
  return acc;
}

__device__ inline void cstore(v8f acc, float* dst, int ldd, int nt, int lane) {
  // C 16x16 f32: vgpr r -> M = r + 8*(lane>=16); N = nt*16 + (lane&15)
  float* p = dst + ((lane & 16) ? 8 : 0) * ldd + nt * 16 + (lane & 15);
#pragma unroll
  for (int r = 0; r < 8; ++r) p[r * ldd] = acc[r];
}

__device__ inline float sigmoidf_(float x) { return 1.f / (1.f + __expf(-x)); }

// elementwise cell update: hn = h + step*sigmoid(gpre+bg)*(tanh(cpre+b)-h)
// result overwrites preC staging (f32).
__device__ inline void cell_stage(const __bf16* h, int ldh, float* preC,
                                  const float* preG, int ld, int Nreal,
                                  const float* vb, int ob, int obg, int ost,
                                  int tid) {
  for (int idx = tid; idx < 16 * Nreal; idx += NTHR) {
    int r = idx / Nreal, c = idx - r * Nreal;
    float hv = (float)h[r * ldh + c];
    float cand = tanhf(preC[r * ld + c] + vb[ob + c]);
    float g = sigmoidf_(preG[r * ld + c] + vb[obg + c]);
    preC[r * ld + c] = hv + vb[ost + c] * g * (cand - hv);
  }
}

// parallel layernorm stats: 16 threads per row, one-pass E[x^2]-m^2,
// wave32 shuffle reduction (16-lane groups are lane-aligned in a wave).
__device__ inline void row_stats(const float* src, int ld, int Nreal,
                                 const float* bias, float* rowm, float* rowr,
                                 int tid) {
  int r = tid >> 4, sub = tid & 15;   // 256 threads -> 16 rows x 16 threads
  float s = 0.f, s2 = 0.f;
  for (int c = sub; c < Nreal; c += 16) {
    float x = src[r * ld + c] + (bias ? bias[c] : 0.f);
    s += x; s2 += x * x;
  }
#pragma unroll
  for (int m = 1; m < 16; m <<= 1) {
    s  += __shfl_xor(s, m, 32);
    s2 += __shfl_xor(s2, m, 32);
  }
  if (sub == 0) {
    float mean = s / (float)Nreal;
    float var = s2 / (float)Nreal - mean * mean;
    rowm[r] = mean;
    rowr[r] = rsqrtf(fmaxf(var, 0.f) + 1e-5f);
  }
}

__device__ inline void ln_write(__bf16* h, int ldh, const float* src, int ld,
                                int Nreal, const float* vb, int olg, int olb,
                                const float* rowm, const float* rowr, int tid) {
  for (int idx = tid; idx < 16 * Nreal; idx += NTHR) {
    int r = idx / Nreal, c = idx - r * Nreal;
    h[r * ldh + c] =
        (__bf16)((src[r * ld + c] - rowm[r]) * rowr[r] * vb[olg + c] + vb[olb + c]);
  }
}

// ===========================================================================
// main recurrent scan kernel: 16 WGs x (8 waves = 256 threads)
// ===========================================================================
__global__ void __launch_bounds__(NTHR)
brain_scan_kernel(const float* __restrict__ X, const __bf16* __restrict__ wb,
                  const float* __restrict__ vb, float* __restrict__ out) {
  __shared__ __align__(16) __bf16 sm_x[16 * 40];
  __shared__ __align__(16) __bf16 sm_xt[16 * 72];
  __shared__ __align__(16) __bf16 sm_s[16 * 72];
  __shared__ __align__(16) __bf16 sm_hp[16 * 72];   // cols 51..63 zero
  __shared__ __align__(16) __bf16 sm_pf[16 * 104];  // cols 90..95 zero
  __shared__ __align__(16) __bf16 sm_cb[16 * 72];   // cols 51..63 zero
  __shared__ float preA[16 * 256];
  __shared__ float preB[16 * 256];
  __shared__ float preC[16 * 96];
  __shared__ float preD[16 * 96];
  __shared__ float rowm[16], rowr[16];

  const int tid = threadIdx.x;
  const int lane = tid & 31;
  // force wave id into an SGPR: job dispatch becomes scalar branches and
  // WMMA always executes with EXEC all-ones (ISA 7.12 requirement).
  const int wvu = __builtin_amdgcn_readfirstlane(tid >> 5);
  const int m0 = blockIdx.x * 16;
  float* hid = out + HID_OFF;

  for (int i = tid; i < 16 * 72; i += NTHR) {
    sm_s[i] = (__bf16)0.f; sm_hp[i] = (__bf16)0.f;
    sm_cb[i] = (__bf16)0.f; sm_xt[i] = (__bf16)0.f;
  }
  for (int i = tid; i < 16 * 104; i += NTHR) sm_pf[i] = (__bf16)0.f;
  __syncthreads();

  for (int t = 0; t < TSTEPS; ++t) {
    // ---- load x_t tile (16 x 32); prefetch next step's tile ----
    for (int idx = tid; idx < 16 * 32; idx += NTHR) {
      int r = idx >> 5, c = idx & 31;
      const float* px = &X[((size_t)(m0 + r) * TSTEPS + t) * DIN + c];
      sm_x[r * 40 + c] = (__bf16)(*px);
      if (t + 1 < TSTEPS) __builtin_prefetch(px + DIN, 0, 3);
    }
    __syncthreads();

    // ---- stage A: input_proj + h2s conn (12 tile jobs) ----
    for (int j = wvu; j < 12; j += 8) {
      v8f acc = {};
      if (j < 4) {
        acc = gacc(acc, sm_x, 40, wb, OFF_IP, 4, j, 1, lane);
        cstore(acc, preA, 64, j, lane);
      } else if (j < 8) {
        int nt = j - 4;
        acc = gacc(acc, sm_hp, 72, wb, OFF_H2S_G, 4, nt, 2, lane);
        cstore(acc, preC, 64, nt, lane);
      } else {
        int nt = j - 8;
        acc = gacc(acc, sm_hp, 72, wb, OFF_H2S_P, 4, nt, 2, lane);
        cstore(acc, preD, 64, nt, lane);
      }
    }
    __syncthreads();
    {
      float nf = (t > 0) ? 1.f : 0.f;
      for (int idx = tid; idx < 16 * 64; idx += NTHR) {
        int r = idx >> 6, c = idx & 63;
        float ip = preA[r * 64 + c] + vb[V_B_IN + c];
        float g = sigmoidf_(preC[r * 64 + c] + vb[V_H2S_GB + c]);
        float p = tanhf(preD[r * 64 + c] + vb[V_H2S_PB + c]);
        sm_xt[r * 72 + c] = (__bf16)(ip + nf * g * p);
      }
    }
    __syncthreads();

    // ---- stage B: sensory (8 jobs) ----
    {
      int j = wvu;  // exactly one job per wave
      int nt = j & 3;
      v8f acc = {};
      if (j < 4) {
        acc = gacc(acc, sm_xt, 72, wb, OFF_SEN_I, 4, nt, 2, lane);
        acc = gacc(acc, sm_s, 72, wb, OFF_SEN_H, 4, nt, 2, lane);
        cstore(acc, preA, 64, nt, lane);
      } else {
        acc = gacc(acc, sm_xt, 72, wb, OFF_SEN_GI, 4, nt, 2, lane);
        acc = gacc(acc, sm_s, 72, wb, OFF_SEN_GH, 4, nt, 2, lane);
        cstore(acc, preB, 64, nt, lane);
      }
    }
    __syncthreads();
    cell_stage(sm_s, 72, preA, preB, 64, 64, vb, V_SEN_B, V_SEN_BG, V_SEN_ST, tid);
    __syncthreads();
    row_stats(preA, 64, 64, nullptr, rowm, rowr, tid);
    __syncthreads();
    ln_write(sm_s, 72, preA, 64, 64, vb, V_SEN_LG, V_SEN_LB, rowm, rowr, tid);
    __syncthreads();

    // ---- stage C: hippocampus (8 jobs; Wh folded into CH/GH) ----
    {
      int j = wvu;
      int nt = j & 3;
      v8f acc = {};
      if (j < 4) {
        acc = gacc(acc, sm_s, 72, wb, OFF_HIP_CI, 4, nt, 2, lane);
        acc = gacc(acc, sm_hp, 72, wb, OFF_HIP_CH, 4, nt, 2, lane);
        cstore(acc, preA, 64, nt, lane);
      } else {
        acc = gacc(acc, sm_s, 72, wb, OFF_HIP_GI, 4, nt, 2, lane);
        acc = gacc(acc, sm_hp, 72, wb, OFF_HIP_GH, 4, nt, 2, lane);
        cstore(acc, preB, 64, nt, lane);
      }
    }
    __syncthreads();
    cell_stage(sm_hp, 72, preA, preB, 64, 51, vb, V_HIP_B, V_HIP_BG, V_HIP_ST, tid);
    __syncthreads();
    row_stats(preA, 64, 51, nullptr, rowm, rowr, tid);
    __syncthreads();
    ln_write(sm_hp, 72, preA, 64, 51, vb, V_HIP_LG, V_HIP_LB, rowm, rowr, tid);
    __syncthreads();

    // ---- stage D: cerebellum (8 jobs; reads OLD pf) ----
    {
      int j = wvu;
      int nt = j & 3;
      v8f acc = {};
      if (j < 4) {
        acc = gacc(acc, sm_s, 72, wb, OFF_CER_CI, 4, nt, 2, lane);
        acc = gacc(acc, sm_pf, 104, wb, OFF_CER_CP, 4, nt, 3, lane);
        acc = gacc(acc, sm_cb, 72, wb, OFF_CER_CH, 4, nt, 2, lane);
        cstore(acc, preA, 64, nt, lane);
      } else {
        acc = gacc(acc, sm_s, 72, wb, OFF_CER_GI, 4, nt, 2, lane);
        acc = gacc(acc, sm_pf, 104, wb, OFF_CER_GP, 4, nt, 3, lane);
        acc = gacc(acc, sm_cb, 72, wb, OFF_CER_GH, 4, nt, 2, lane);
        cstore(acc, preB, 64, nt, lane);
      }
    }
    __syncthreads();
    cell_stage(sm_cb, 72, preA, preB, 64, 51, vb, V_CER_B, V_CER_BG, V_CER_ST, tid);
    __syncthreads();
    row_stats(preA, 64, 51, nullptr, rowm, rowr, tid);
    __syncthreads();
    ln_write(sm_cb, 72, preA, 64, 51, vb, V_CER_LG, V_CER_LB, rowm, rowr, tid);
    __syncthreads();

    // ---- stage E: prefrontal + c2p (24 jobs; pf read is OLD pf) ----
    for (int j = wvu; j < 24; j += 8) {
      int nt = j % 6;
      v8f acc = {};
      if (j < 6) {
        acc = gacc(acc, sm_s, 72, wb, OFF_PRE_CS, 6, nt, 2, lane);
        acc = gacc(acc, sm_hp, 72, wb, OFF_PRE_CH, 6, nt, 2, lane);
        acc = gacc(acc, sm_cb, 72, wb, OFF_PRE_CB, 6, nt, 2, lane);
        acc = gacc(acc, sm_pf, 104, wb, OFF_PRE_CF, 6, nt, 3, lane);
        cstore(acc, preA, 96, nt, lane);
      } else if (j < 12) {
        acc = gacc(acc, sm_s, 72, wb, OFF_PRE_GS, 6, nt, 2, lane);
        acc = gacc(acc, sm_hp, 72, wb, OFF_PRE_GH, 6, nt, 2, lane);
        acc = gacc(acc, sm_cb, 72, wb, OFF_PRE_GB, 6, nt, 2, lane);
        acc = gacc(acc, sm_pf, 104, wb, OFF_PRE_GF, 6, nt, 3, lane);
        cstore(acc, preB, 96, nt, lane);
      } else if (j < 18) {
        acc = gacc(acc, sm_cb, 72, wb, OFF_C2P_G, 6, nt, 2, lane);
        cstore(acc, preC, 96, nt, lane);
      } else {
        acc = gacc(acc, sm_cb, 72, wb, OFF_C2P_P, 6, nt, 2, lane);
        cstore(acc, preD, 96, nt, lane);
      }
    }
    __syncthreads();
    cell_stage(sm_pf, 104, preA, preB, 96, 90, vb, V_PRE_B, V_PRE_BG, V_PRE_ST, tid);
    __syncthreads();
    row_stats(preA, 96, 90, nullptr, rowm, rowr, tid);
    __syncthreads();
    for (int idx = tid; idx < 16 * 90; idx += NTHR) {
      int r = idx / 90, c = idx - r * 90;
      float lnv = (preA[r * 96 + c] - rowm[r]) * rowr[r] * vb[V_PRE_LG + c] + vb[V_PRE_LB + c];
      float corr = sigmoidf_(preC[r * 96 + c] + vb[V_C2P_GB + c]) *
                   tanhf(preD[r * 96 + c] + vb[V_C2P_PB + c]);
      sm_pf[r * 104 + c] = (__bf16)(lnv + 0.1f * corr);
    }
    __syncthreads();

    // ---- stage F: thalamic gate / proj (32 jobs, N = 256) ----
    for (int j = wvu; j < 32; j += 8) {
      int nt = j & 15;
      v8f acc = {};
      if (j < 16) {
        acc = gacc(acc, sm_s, 72, wb, OFF_THL_S, 16, nt, 2, lane);
        acc = gacc(acc, sm_hp, 72, wb, OFF_THL_H, 16, nt, 2, lane);
        acc = gacc(acc, sm_pf, 104, wb, OFF_THL_P, 16, nt, 3, lane);
        acc = gacc(acc, sm_cb, 72, wb, OFF_THL_C, 16, nt, 2, lane);
        cstore(acc, preA, 256, nt, lane);
      } else {
        acc = gacc(acc, sm_s, 72, wb, OFF_THP_S, 16, nt, 2, lane);
        acc = gacc(acc, sm_hp, 72, wb, OFF_THP_H, 16, nt, 2, lane);
        acc = gacc(acc, sm_pf, 104, wb, OFF_THP_P, 16, nt, 3, lane);
        acc = gacc(acc, sm_cb, 72, wb, OFF_THP_C, 16, nt, 2, lane);
        cstore(acc, preB, 256, nt, lane);
      }
    }
    __syncthreads();
    row_stats(preA, 256, 256, vb + V_THL_B, rowm, rowr, tid);
    __syncthreads();
    for (int idx = tid; idx < 16 * 256; idx += NTHR) {
      int r = idx >> 8, c = idx & 255;
      float a = preA[r * 256 + c] + vb[V_THL_B + c];
      float g = sigmoidf_((a - rowm[r]) * rowr[r] * vb[V_TH_LG + c] + vb[V_TH_LB + c]);
      float pj = preB[r * 256 + c] + vb[V_THP_B + c];
      hid[((size_t)(m0 + r) * TSTEPS + t) * DH + c] = g * pj;
    }
    __syncthreads();
  }
}

// ===========================================================================
// head kernels
// ===========================================================================
__global__ void __launch_bounds__(256)
head_pred_kernel(const float* __restrict__ hid, const float* __restrict__ X,
                 const float* __restrict__ lng, const float* __restrict__ lnb,
                 const float* __restrict__ W, const float* __restrict__ bias,
                 float* __restrict__ pred, float* __restrict__ surp) {
  __shared__ float hn[8][256];
  int lane = threadIdx.x & 31, wvv = threadIdx.x >> 5;
  long long row = (long long)blockIdx.x * 8 + wvv;
  if (row >= 256LL * 511) return;
  int b = (int)(row / 511), tt = (int)(row % 511);
  const float* hr = hid + ((size_t)b * 512 + tt) * 256;
  float v[8], s = 0.f;
#pragma unroll
  for (int i = 0; i < 8; ++i) { v[i] = hr[i * 32 + lane]; s += v[i]; }
#pragma unroll
  for (int m = 16; m > 0; m >>= 1) s += __shfl_xor(s, m, 32);
  float mean = s * (1.f / 256.f);
  float q = 0.f;
#pragma unroll
  for (int i = 0; i < 8; ++i) { float d = v[i] - mean; q += d * d; }
#pragma unroll
  for (int m = 16; m > 0; m >>= 1) q += __shfl_xor(q, m, 32);
  float rstd = rsqrtf(q * (1.f / 256.f) + 1e-5f);
#pragma unroll
  for (int i = 0; i < 8; ++i) {
    int k = i * 32 + lane;
    hn[wvv][k] = (v[i] - mean) * rstd * lng[k] + lnb[k];
  }
  float acc = bias[lane];
  for (int k = 0; k < 256; ++k) acc += hn[wvv][k] * W[k * 32 + lane];
  pred[row * 32 + lane] = acc;
  float d = acc - X[((size_t)b * 512 + (tt + 1)) * 32 + lane];
  float e = d * d;
#pragma unroll
  for (int m = 16; m > 0; m >>= 1) e += __shfl_xor(e, m, 32);
  if (lane == 0) surp[row] = e * (1.f / 32.f);
}

__global__ void __launch_bounds__(256)
head_anom_kernel(const float* __restrict__ hid, const float* __restrict__ lng,
                 const float* __restrict__ lnb, const float* __restrict__ W,
                 const float* __restrict__ bias, float* __restrict__ anom) {
  int lane = threadIdx.x & 31, wvv = threadIdx.x >> 5;
  long long row = (long long)blockIdx.x * 8 + wvv;
  if (row >= 256LL * 512) return;
  const float* hr = hid + (size_t)row * 256;
  float v[8], s = 0.f;
#pragma unroll
  for (int i = 0; i < 8; ++i) { v[i] = hr[i * 32 + lane]; s += v[i]; }
#pragma unroll
  for (int m = 16; m > 0; m >>= 1) s += __shfl_xor(s, m, 32);
  float mean = s * (1.f / 256.f);
  float q = 0.f;
#pragma unroll
  for (int i = 0; i < 8; ++i) { float d = v[i] - mean; q += d * d; }
#pragma unroll
  for (int m = 16; m > 0; m >>= 1) q += __shfl_xor(q, m, 32);
  float rstd = rsqrtf(q * (1.f / 256.f) + 1e-5f);
  float acc = 0.f;
#pragma unroll
  for (int i = 0; i < 8; ++i) {
    int k = i * 32 + lane;
    acc += ((v[i] - mean) * rstd * lng[k] + lnb[k]) * W[k];
  }
#pragma unroll
  for (int m = 16; m > 0; m >>= 1) acc += __shfl_xor(acc, m, 32);
  if (lane == 0) anom[row] = 1.f / (1.f + __expf(-(acc + bias[0])));
}

__global__ void __launch_bounds__(128)
head_prop_kernel(const float* __restrict__ hid, const float* __restrict__ lng,
                 const float* __restrict__ lnb, const float* __restrict__ W1,
                 const float* __restrict__ b1, const float* __restrict__ W2,
                 const float* __restrict__ b2, float* __restrict__ prop) {
  __shared__ float hn[256];
  __shared__ float u[128];
  __shared__ float part[128];
  int t = threadIdx.x, b = blockIdx.x;
  const float* hr = hid + ((size_t)b * 512 + 511) * 256;
  float a0 = hr[t], a1 = hr[t + 128];
  part[t] = a0 + a1;
  __syncthreads();
  for (int off = 64; off > 0; off >>= 1) {
    if (t < off) part[t] += part[t + off];
    __syncthreads();
  }
  float mean = part[0] * (1.f / 256.f);
  __syncthreads();
  float d0 = a0 - mean, d1 = a1 - mean;
  part[t] = d0 * d0 + d1 * d1;
  __syncthreads();
  for (int off = 64; off > 0; off >>= 1) {
    if (t < off) part[t] += part[t + off];
    __syncthreads();
  }
  float rstd = rsqrtf(part[0] * (1.f / 256.f) + 1e-5f);
  hn[t] = d0 * rstd * lng[t] + lnb[t];
  hn[t + 128] = d1 * rstd * lng[t + 128] + lnb[t + 128];
  __syncthreads();
  float acc = b1[t];
  for (int k = 0; k < 256; ++k) acc += hn[k] * W1[k * 128 + t];
  u[t] = 0.5f * acc * (1.f + erff(acc * 0.70710678118654752f));  // exact gelu
  __syncthreads();
  if (t < 8) {
    float o = b2[t];
    for (int k = 0; k < 128; ++k) o += u[k] * W2[k * 8 + t];
    prop[b * 8 + t] = o;
  }
}

// ===========================================================================
// launch
// ===========================================================================
extern "C" void kernel_launch(void* const* d_in, const int* in_sizes, int n_in,
                              void* d_out, int out_size, void* d_ws, size_t ws_size,
                              hipStream_t stream) {
  (void)in_sizes; (void)out_size; (void)ws_size;
  P67 P;
  for (int i = 0; i < 67 && i < n_in; ++i) P.q[i] = (const float*)d_in[i];
  __bf16* wbuf = (__bf16*)d_ws;
  float* vbuf = (float*)((char*)d_ws + VOFF_BYTES);
  float* out = (float*)d_out;

  repack_kernel<<<36, 256, 0, stream>>>(P, wbuf, vbuf);
  brain_scan_kernel<<<16, NTHR, 0, stream>>>(P.q[0], wbuf, vbuf, out);
  head_pred_kernel<<<(int)((256LL * 511 + 7) / 8), 256, 0, stream>>>(
      out + HID_OFF, P.q[0], P.q[53], P.q[54], P.q[55], P.q[56],
      out + PRED_OFF, out + SURP_OFF);
  head_anom_kernel<<<(int)((256LL * 512 + 7) / 8), 256, 0, stream>>>(
      out + HID_OFF, P.q[57], P.q[58], P.q[59], P.q[60], out + ANOM_OFF);
  head_prop_kernel<<<256, 128, 0, stream>>>(
      out + HID_OFF, P.q[61], P.q[62], P.q[63], P.q[64], P.q[65], P.q[66],
      out + PROP_OFF);
}
